// MAMNetModule_35493609734578
// MI455X (gfx1250) — matrix operand order repaired and checked by reading, not compiled, not run on hardware
//
#include <hip/hip_runtime.h>
#include <hip/hip_bf16.h>
#include <cstdint>
#include <type_traits>

typedef __attribute__((ext_vector_type(16))) _Float16 v16h;
typedef __attribute__((ext_vector_type(8)))  _Float16 v8h;
typedef __attribute__((ext_vector_type(8)))  float    v8f;

// gcc-style int4 vector: matches the async-load builtin's parameter type
typedef int v4i __attribute__((vector_size(16)));
typedef __attribute__((address_space(1))) v4i g_v4i;   // global int4
typedef __attribute__((address_space(3))) v4i l_v4i;   // LDS int4

#define TPW 4                 // pixel tiles (16 px each) per wave
#define WROW 584              // padded LDS row stride (halves) for 16x576 tile

// ---------------------------------------------------------------------------
// Implicit-GEMM 3x3 conv, stride 1, pad 1, Cin = 64, square images.
//   input : NHWC f16  [B][HW][HW][64]
//   w2    : f16       [CoutPad][9][64]   (K = kidx*64 + ci)
//   bias  : f32       [CoutReal]
//   out   : NHWC f16  [B][HW][HW][CoutPad]
//   res   : optional residual (NHWC f16, same shape as out)
// 256-thread block (8 waves). The block's 16(Cout)x576(K) f16 weight tile is
// staged into LDS once via async load-to-LDS; each wave computes 4 independent
// 16x16 output tiles -> 72 v_wmma_f32_16x16x32_f16 per wave. Interior tiles
// (no border taps) run an unchecked loop with immediate-offset B loads.
// Requires HW*(HW/16) % 32 == 0 (true for 96/192/384).
// ---------------------------------------------------------------------------
template<int HW, int CoutPad>
__global__ __launch_bounds__(256)
void conv3_wmma(const _Float16* __restrict__ in, const _Float16* __restrict__ w2,
                const float* __restrict__ bias, _Float16* __restrict__ out,
                const _Float16* __restrict__ res, int CoutReal, int relu)
{
    constexpr int H = HW, W = HW;
    constexpr int tilesX = W / 16;

    __shared__ _Float16 sw[16 * WROW];

    const int tid  = threadIdx.x;
    const int lane = tid & 31;
    const int wave = tid >> 5;
    const int m    = lane & 15;      // A-row / B,C column (N)
    const int hi   = lane >> 4;      // lane group
    const int coB  = blockIdx.y << 4;
    const int b    = blockIdx.z;

    // ---- stage 16 x 576 f16 weight rows into LDS (72 b128 chunks per row) ----
    const _Float16* wg = w2 + (size_t)coB * 576;
#if __has_builtin(__builtin_amdgcn_global_load_async_to_lds_b128)
    for (int c = tid; c < 16 * 72; c += 256) {
        const int co = c / 72;
        const int k8 = (c % 72) * 8;
        __builtin_amdgcn_global_load_async_to_lds_b128(
            (g_v4i*)(uintptr_t)(wg + (size_t)co * 576 + k8),
            (l_v4i*)(unsigned)(uintptr_t)(sw + co * WROW + k8),
            0, 0);
    }
  #if __has_builtin(__builtin_amdgcn_s_wait_asynccnt)
    __builtin_amdgcn_s_wait_asynccnt(0);
  #else
    asm volatile("s_wait_asynccnt 0x0" ::: "memory");
  #endif
#else
    for (int c = tid; c < 16 * 72; c += 256) {
        const int co = c / 72;
        const int k8 = (c % 72) * 8;
        *(v8h*)(sw + co * WROW + k8) = *(const v8h*)(wg + (size_t)co * 576 + k8);
    }
#endif
    __syncthreads();

    const int tile0 = (blockIdx.x * 8 + wave) * TPW;

    const _Float16* inB  = in  + (size_t)b * H * W * 64;
    _Float16*       outB = out + (size_t)b * H * W * CoutPad;

    int tyA[TPW], txA[TPW];
    const _Float16* pb[TPW];
    bool allInt = true;
    #pragma unroll
    for (int t = 0; t < TPW; ++t) {
        const int tl = tile0 + t;
        const int tY = tl / tilesX;          // constant divisor
        const int tX = tl - tY * tilesX;
        tyA[t] = tY;
        txA[t] = tX * 16 + m;                // this lane's pixel (N = lane&15)
        pb[t]  = inB + ((size_t)tY * W + tX * 16 + m) * 64 + hi * 16;
        allInt = allInt && (tY >= 1) && (tY <= H - 2) &&
                           (tX >= 1) && (tX <= tilesX - 2);
    }

    const v8f zero8 = {};
    v8f acc[TPW];
    #pragma unroll
    for (int t = 0; t < TPW; ++t) acc[t] = zero8;

    auto body = [&](auto chk) {
        constexpr bool CHK = decltype(chk)::value;
        #pragma unroll
        for (int kidx = 0; kidx < 9; ++kidx) {
            const int dy = kidx / 3 - 1;
            const int dx = kidx % 3 - 1;
            const _Float16* swp = sw + m * WROW + kidx * 64 + hi * 16;
            #pragma unroll
            for (int half = 0; half < 2; ++half) {    // ci block 0..31 / 32..63
                // A fragment (contiguous k order, same as B):
                //  lanes<16: halves0-15 -> K=k0..+15
                //  lanes>=16 shift K by +16
                v8h alo = *(const v8h*)(swp + half * 32);
                v8h ahi = *(const v8h*)(swp + half * 32 + 8);
                v16h a;
                #pragma unroll
                for (int e = 0; e < 8; ++e) { a[e] = alo[e]; a[e + 8] = ahi[e]; }
                #pragma unroll
                for (int t = 0; t < TPW; ++t) {
                    // B fragment (32x16): lane N = lane&15, half e -> K=e+16*hi
                    v16h bf;
                    if constexpr (CHK) {
                        #pragma unroll
                        for (int e = 0; e < 16; ++e) bf[e] = (_Float16)0.f;
                        const int yy = tyA[t] + dy;
                        const int xx = txA[t] + dx;
                        if (yy >= 0 && yy < H && xx >= 0 && xx < W)
                            bf = *(const v16h*)(inB + ((size_t)yy * W + xx) * 64 +
                                                half * 32 + hi * 16);
                    } else {
                        // compile-time displacement -> immediate-offset loads
                        bf = *(const v16h*)(pb[t] + (dy * W + dx) * 64 + half * 32);
                    }
                    acc[t] = __builtin_amdgcn_wmma_f32_16x16x32_f16(
                        false, a, false, bf, (short)0, acc[t], false, false);
                }
            }
        }
    };
    if (allInt) body(std::integral_constant<bool, false>{});
    else        body(std::integral_constant<bool, true>{});

    // D layout: VGPR r, lane L -> M = r + 8*(L>=16), N = L&15
    float bv[8];
    #pragma unroll
    for (int r = 0; r < 8; ++r) {
        const int co = coB + hi * 8 + r;
        bv[r] = (co < CoutReal) ? bias[co] : 0.f;
    }
    #pragma unroll
    for (int t = 0; t < TPW; ++t) {
        const size_t obase = ((size_t)tyA[t] * W + txA[t]) * CoutPad + coB + hi * 8;
        float radd[8];
        #pragma unroll
        for (int r = 0; r < 8; ++r) radd[r] = 0.f;
        if (res) {
            v8h rv = *(const v8h*)(res + (size_t)b * H * W * CoutPad + obase);
            #pragma unroll
            for (int r = 0; r < 8; ++r) radd[r] = (float)rv[r];
        }
        v8h o;
        #pragma unroll
        for (int r = 0; r < 8; ++r) {
            float v = acc[t][r] + bv[r];
            if (relu) v = fmaxf(v, 0.f);
            v += radd[r];
            o[r] = (_Float16)v;
        }
        *(v8h*)(outB + obase) = o;
    }
}

template<int HW, int CoP>
static void conv_launch(const _Float16* in, const _Float16* w2, const float* bias,
                        _Float16* out, const _Float16* res, int coR, int relu,
                        hipStream_t stream)
{
    dim3 g((HW * (HW / 16)) / 32, CoP / 16, 4);      // 8 waves x 4 tiles / block
    conv3_wmma<HW, CoP><<<g, dim3(256), 0, stream>>>(in, w2, bias, out, res, coR, relu);
}

// ---------------------------------------------------------------------------
// Weight swizzle: f32 [CoutReal][64][3][3] -> f16 [CoutPad][9][64]
// ---------------------------------------------------------------------------
__global__ void prep_w(const float* __restrict__ w, _Float16* __restrict__ w2,
                       int CoutReal, int CoutPad)
{
    const int idx = blockIdx.x * 256 + threadIdx.x;
    const int total = CoutPad * 576;
    if (idx >= total) return;
    const int co = idx / 576;
    const int r  = idx % 576;
    const int kidx = r / 64;
    const int ci   = r % 64;
    const float v = (co < CoutReal) ? w[((size_t)co * 64 + ci) * 9 + kidx] : 0.f;
    w2[idx] = (_Float16)v;
}

// ---------------------------------------------------------------------------
// First conv: 3->64 channels, NCHW f32 in, NHWC f16 out (written to h0 and t)
// ---------------------------------------------------------------------------
__global__ void first_conv(const float* __restrict__ x, const float* __restrict__ w,
                           const float* __restrict__ bias,
                           _Float16* __restrict__ h0, _Float16* __restrict__ t)
{
    const int idx = blockIdx.x * 256 + threadIdx.x;   // B*96*96*64
    if (idx >= 4 * 96 * 96 * 64) return;
    const int co = idx & 63;
    int tt = idx >> 6;
    const int xx = tt % 96; tt /= 96;
    const int yy = tt % 96;
    const int b  = tt / 96;
    float acc = bias[co];
    #pragma unroll
    for (int ci = 0; ci < 3; ++ci) {
        #pragma unroll
        for (int k = 0; k < 9; ++k) {
            const int ky = k / 3 - 1, kx = k % 3 - 1;
            const int u = yy + ky, v = xx + kx;
            if (u >= 0 && u < 96 && v >= 0 && v < 96)
                acc += x[(((size_t)b * 3 + ci) * 96 + u) * 96 + v] *
                       w[((size_t)co * 3 + ci) * 9 + k];
        }
    }
    h0[idx] = (_Float16)acc;
    t[idx]  = (_Float16)acc;
}

// ---------------------------------------------------------------------------
// MAM attention stats: per-channel spatial variance (ddof=1), normalized over
// channels (mean / std ddof=1), gate = 0.5*sigmoid(z). gates: [B][64]
// ---------------------------------------------------------------------------
__global__ __launch_bounds__(256)
void mam_stats(const _Float16* __restrict__ r1, float* __restrict__ gates, int HW)
{
    __shared__ float sh_s[256];
    __shared__ float sh_q[256];
    __shared__ float chvar[64];
    __shared__ float mm[2];
    const int b   = blockIdx.x;
    const int c   = threadIdx.x & 63;
    const int seg = threadIdx.x >> 6;
    const _Float16* p = r1 + (size_t)b * HW * 64 + c;
    float s = 0.f, q = 0.f;
    for (int i = seg; i < HW; i += 4) {
        const float v = (float)p[(size_t)i * 64];
        s += v; q += v * v;
    }
    sh_s[threadIdx.x] = s; sh_q[threadIdx.x] = q;
    __syncthreads();
    if (seg == 0) {
        for (int k = 1; k < 4; ++k) { s += sh_s[c + (k << 6)]; q += sh_q[c + (k << 6)]; }
        const float mean = s / (float)HW;
        chvar[c] = (q - s * mean) / (float)(HW - 1);
    }
    __syncthreads();
    if (threadIdx.x == 0) {
        float mu = 0.f;
        for (int k = 0; k < 64; ++k) mu += chvar[k];
        mu *= (1.f / 64.f);
        float sd = 0.f;
        for (int k = 0; k < 64; ++k) { const float d = chvar[k] - mu; sd += d * d; }
        mm[0] = mu; mm[1] = sqrtf(sd / 63.f);
    }
    __syncthreads();
    if (seg == 0) {
        const float z = (chvar[c] - mm[0]) / mm[1];
        gates[b * 64 + c] = 0.5f / (1.f + __expf(-z));
    }
}

// t[:,c] += gate(src)*r1[:,src]; src = c+16 (c<48) else c-48; +0.5 for src<16
__global__ void mam_apply(_Float16* __restrict__ t, const _Float16* __restrict__ r1,
                          const float* __restrict__ gates, int HW)
{
    const size_t idx = (size_t)blockIdx.x * 256 + threadIdx.x;
    if (idx >= (size_t)4 * HW * 64) return;
    const int c = (int)(idx & 63);
    const size_t pixbase = idx - c;
    const int b = (int)((idx >> 6) / (size_t)HW);
    const int src = (c < 48) ? c + 16 : c - 48;
    const float g = gates[b * 64 + src] + ((src < 16) ? 0.5f : 0.f);
    t[idx] = (_Float16)((float)t[idx] + g * (float)r1[pixbase + src]);
}

// ---------------------------------------------------------------------------
// Pixel shuffle r=2 (NHWC): [b][h][w][256] -> [b][2h][2w][64]
// ---------------------------------------------------------------------------
__global__ void pshuf(const _Float16* __restrict__ in, _Float16* __restrict__ out,
                      int h, int w)
{
    const size_t total = (size_t)4 * (2 * h) * (2 * w) * 64;
    const size_t idx = (size_t)blockIdx.x * 256 + threadIdx.x;
    if (idx >= total) return;
    const int co = (int)(idx & 63);
    size_t t1 = idx >> 6;
    const int ox = (int)(t1 % (2 * w)); t1 /= (2 * w);
    const int oy = (int)(t1 % (2 * h));
    const int b  = (int)(t1 / (2 * h));
    out[idx] = in[(((size_t)b * h + (oy >> 1)) * w + (ox >> 1)) * 256 +
                  co * 4 + ((oy & 1) << 1) + (ox & 1)];
}

// ---------------------------------------------------------------------------
// Bilinear 4x resize of x (96 -> 384, NCHW f32, half-pixel, clamped).
// Writes d_out (base term of the final sum) and a workspace copy.
// ---------------------------------------------------------------------------
__global__ void resize_base(const float* __restrict__ x, float* __restrict__ outp,
                            float* __restrict__ basecpy)
{
    const int idx = blockIdx.x * 256 + threadIdx.x;   // 4*3*384*384
    if (idx >= 4 * 3 * 384 * 384) return;
    const int ox = idx % 384;
    int t = idx / 384;
    const int oy = t % 384;
    const int p  = t / 384;                           // b*3 + c
    const float fy = (oy + 0.5f) * 0.25f - 0.5f;
    const float fx = (ox + 0.5f) * 0.25f - 0.5f;
    const int y0 = (int)floorf(fy); const float wy = fy - y0;
    const int x0 = (int)floorf(fx); const float wx = fx - x0;
    const int y0c = min(max(y0, 0), 95),     y1c = min(max(y0 + 1, 0), 95);
    const int x0c = min(max(x0, 0), 95),     x1c = min(max(x0 + 1, 0), 95);
    const float* s = x + (size_t)p * 96 * 96;
    const float v = (1.f - wy) * ((1.f - wx) * s[y0c * 96 + x0c] + wx * s[y0c * 96 + x1c]) +
                    wy         * ((1.f - wx) * s[y1c * 96 + x0c] + wx * s[y1c * 96 + x1c]);
    outp[idx] = v;
    basecpy[idx] = v;
}

// Second 4x bilinear (384 -> 1536) but only the top-left 385x385 slab is ever
// read by pixel_conv, so compute just that. slab: [b*3+c][385][385] f32
__global__ void resize_slab(const float* __restrict__ base, float* __restrict__ slab)
{
    const int idx = blockIdx.x * 256 + threadIdx.x;   // 12*385*385
    if (idx >= 12 * 385 * 385) return;
    const int ox = idx % 385;
    int t = idx / 385;
    const int oy = t % 385;
    const int p  = t / 385;
    const float fy = (oy + 0.5f) * 0.25f - 0.5f;
    const float fx = (ox + 0.5f) * 0.25f - 0.5f;
    const int y0 = (int)floorf(fy); const float wy = fy - y0;
    const int x0 = (int)floorf(fx); const float wx = fx - x0;
    const int y0c = min(max(y0, 0), 383),    y1c = min(max(y0 + 1, 0), 383);
    const int x0c = min(max(x0, 0), 383),    x1c = min(max(x0 + 1, 0), 383);
    const float* s = base + (size_t)p * 384 * 384;
    slab[idx] = (1.f - wy) * ((1.f - wx) * s[y0c * 384 + x0c] + wx * s[y0c * 384 + x1c]) +
                wy         * ((1.f - wx) * s[y1c * 384 + x0c] + wx * s[y1c * 384 + x1c]);
}

// out1: per-pixel dynamic 3x3 kernels; ker NHWC f16 [b][384][384][32] (27 used)
// channel (n*3 + c), tap n = (dx+1)*3 + (dy+1); accumulates into d_out (NCHW)
__global__ void pconv_apply(const float* __restrict__ slab,
                            const _Float16* __restrict__ ker, float* __restrict__ outp)
{
    const int idx = blockIdx.x * 256 + threadIdx.x;   // 4*3*384*384
    if (idx >= 4 * 3 * 384 * 384) return;
    const int xq = idx % 384;
    int t = idx / 384;
    const int yq = t % 384; t /= 384;
    const int c  = t % 3;
    const int b  = t / 3;
    const _Float16* kp = ker + (((size_t)b * 384 + yq) * 384 + xq) * 32;
    const float* sl = slab + (size_t)(b * 3 + c) * 385 * 385;
    float acc = outp[idx];
    #pragma unroll
    for (int n = 0; n < 9; ++n) {
        const int dx = n / 3 - 1, dy = n % 3 - 1;
        const int u = yq + dx, v = xq + dy;
        const float pv = (u >= 0 && v >= 0) ? sl[(size_t)u * 385 + v] : 0.f;
        acc += pv * (float)kp[n * 3 + c];
    }
    outp[idx] = acc;
}

// out2: add final-conv result (NHWC f16 pad16) into d_out (NCHW f32)
__global__ void add_final(const _Float16* __restrict__ fin, float* __restrict__ outp)
{
    const int idx = blockIdx.x * 256 + threadIdx.x;   // 4*3*384*384
    if (idx >= 4 * 3 * 384 * 384) return;
    const int xq = idx % 384;
    int t = idx / 384;
    const int yq = t % 384; t /= 384;
    const int c  = t % 3;
    const int b  = t / 3;
    outp[idx] += (float)fin[(((size_t)b * 384 + yq) * 384 + xq) * 16 + c];
}

// ---------------------------------------------------------------------------
extern "C" void kernel_launch(void* const* d_in, const int* in_sizes, int n_in,
                              void* d_out, int out_size, void* d_ws, size_t ws_size,
                              hipStream_t stream) {
    (void)in_sizes; (void)n_in; (void)out_size; (void)ws_size;
    const float* x       = (const float*)d_in[0];
    const float* w_first = (const float*)d_in[1];
    const float* b_first = (const float*)d_in[2];
    const float* w_blk1  = (const float*)d_in[3];
    const float* b_blk1  = (const float*)d_in[4];
    const float* w_blk2  = (const float*)d_in[5];
    const float* b_blk2  = (const float*)d_in[6];
    const float* w_after = (const float*)d_in[7];
    const float* b_after = (const float*)d_in[8];
    const float* w_up1   = (const float*)d_in[9];
    const float* b_up1   = (const float*)d_in[10];
    const float* w_up2   = (const float*)d_in[11];
    const float* b_up2   = (const float*)d_in[12];
    const float* w_fc1   = (const float*)d_in[13];
    const float* b_fc1   = (const float*)d_in[14];
    const float* w_fc2   = (const float*)d_in[15];
    const float* b_fc2   = (const float*)d_in[16];
    const float* w_kconv = (const float*)d_in[17];
    const float* b_kconv = (const float*)d_in[18];
    const float* w_final = (const float*)d_in[19];
    const float* b_final = (const float*)d_in[20];
    float* outp = (float*)d_out;

    char* ws = (char*)d_ws;
    size_t off = 0;
    auto alloc = [&](size_t bytes) -> char* {
        char* p = ws + off;
        off = (off + bytes + 255) & ~(size_t)255;
        return p;
    };

    _Float16* h0     = (_Float16*)alloc((size_t)4 * 96 * 96 * 64 * 2);
    _Float16* tbuf   = (_Float16*)alloc((size_t)4 * 96 * 96 * 64 * 2);
    _Float16* tmp1   = (_Float16*)alloc((size_t)4 * 96 * 96 * 64 * 2);
    _Float16* tmp2   = (_Float16*)alloc((size_t)4 * 96 * 96 * 64 * 2);
    _Float16* up1out = (_Float16*)alloc((size_t)4 * 96 * 96 * 256 * 2);
    _Float16* s1     = (_Float16*)alloc((size_t)4 * 192 * 192 * 64 * 2);
    _Float16* Dbuf   = (_Float16*)alloc((size_t)4 * 192 * 192 * 256 * 2);  // also fc1 out
    _Float16* Ebuf   = (_Float16*)alloc((size_t)4 * 384 * 384 * 64 * 2);   // s2, then f
    _Float16* kout   = (_Float16*)alloc((size_t)4 * 384 * 384 * 32 * 2);
    _Float16* finout = (_Float16*)alloc((size_t)4 * 384 * 384 * 16 * 2);
    float*    wsbase = (float*)alloc((size_t)4 * 3 * 384 * 384 * 4);
    float*    slab   = (float*)alloc((size_t)12 * 385 * 385 * 4);
    float*    gates  = (float*)alloc((size_t)4 * 64 * 4);
    _Float16* wB1    = (_Float16*)alloc((size_t)8 * 64 * 576 * 2);
    _Float16* wB2    = (_Float16*)alloc((size_t)8 * 64 * 576 * 2);
    _Float16* wAf    = (_Float16*)alloc((size_t)64 * 576 * 2);
    _Float16* wU1    = (_Float16*)alloc((size_t)256 * 576 * 2);
    _Float16* wU2    = (_Float16*)alloc((size_t)256 * 576 * 2);
    _Float16* wF1    = (_Float16*)alloc((size_t)64 * 576 * 2);
    _Float16* wF2    = (_Float16*)alloc((size_t)64 * 576 * 2);
    _Float16* wKc    = (_Float16*)alloc((size_t)32 * 576 * 2);
    _Float16* wFn    = (_Float16*)alloc((size_t)16 * 576 * 2);

    auto prep = [&](const float* w, _Float16* w2, int coR, int coP) {
        const int total = coP * 576;
        prep_w<<<dim3((total + 255) / 256), dim3(256), 0, stream>>>(w, w2, coR, coP);
    };
    for (int i = 0; i < 8; ++i) {
        prep(w_blk1 + (size_t)i * 64 * 64 * 9, wB1 + (size_t)i * 64 * 576, 64, 64);
        prep(w_blk2 + (size_t)i * 64 * 64 * 9, wB2 + (size_t)i * 64 * 576, 64, 64);
    }
    prep(w_after, wAf, 64, 64);
    prep(w_up1,   wU1, 256, 256);
    prep(w_up2,   wU2, 256, 256);
    prep(w_fc1,   wF1, 64, 64);
    prep(w_fc2,   wF2, 64, 64);
    prep(w_kconv, wKc, 27, 32);
    prep(w_final, wFn, 3, 16);

    // base term (writes all of d_out) + 385x385 slab of the 16x bilinear
    resize_base<<<dim3((4 * 3 * 384 * 384 + 255) / 256), dim3(256), 0, stream>>>(x, outp, wsbase);
    resize_slab<<<dim3((12 * 385 * 385 + 255) / 256), dim3(256), 0, stream>>>(wsbase, slab);

    // first conv 3->64, writes h0 and t
    first_conv<<<dim3((4 * 96 * 96 * 64 + 255) / 256), dim3(256), 0, stream>>>(
        x, w_first, b_first, h0, tbuf);

    // 8 MAM blocks at 96x96
    for (int i = 0; i < 8; ++i) {
        conv_launch<96, 64>(tbuf, wB1 + (size_t)i * 64 * 576, b_blk1 + i * 64,
                            tmp1, nullptr, 64, 1, stream);
        conv_launch<96, 64>(tmp1, wB2 + (size_t)i * 64 * 576, b_blk2 + i * 64,
                            tmp2, nullptr, 64, 0, stream);
        mam_stats<<<dim3(4), dim3(256), 0, stream>>>(tmp2, gates, 96 * 96);
        mam_apply<<<dim3((4 * 96 * 96 * 64 + 255) / 256), dim3(256), 0, stream>>>(
            tbuf, tmp2, gates, 96 * 96);
    }

    // t = h0 + conv_after(t)
    conv_launch<96, 64>(tbuf, wAf, b_after, tmp1, h0, 64, 0, stream);

    // upsample stage 1: conv 64->256 @96, shuffle -> 64 @192
    conv_launch<96, 256>(tmp1, wU1, b_up1, up1out, nullptr, 256, 0, stream);
    pshuf<<<dim3((int)(((size_t)4 * 192 * 192 * 64 + 255) / 256)), dim3(256), 0, stream>>>(
        up1out, s1, 96, 96);

    // upsample stage 2: conv 64->256 @192, shuffle -> 64 @384
    conv_launch<192, 256>(s1, wU2, b_up2, Dbuf, nullptr, 256, 0, stream);
    pshuf<<<dim3((int)(((size_t)4 * 384 * 384 * 64 + 255) / 256)), dim3(256), 0, stream>>>(
        Dbuf, Ebuf, 192, 192);

    // fc1, fc2 @384 (ping-pong Ebuf <-> Dbuf; f ends in Ebuf)
    conv_launch<384, 64>(Ebuf, wF1, b_fc1, Dbuf, nullptr, 64, 0, stream);
    conv_launch<384, 64>(Dbuf, wF2, b_fc2, Ebuf, nullptr, 64, 0, stream);

    // dynamic-kernel predictor (27ch, padded 32) and final conv (3ch, padded 16)
    conv_launch<384, 32>(Ebuf, wKc, b_kconv, kout,   nullptr, 27, 0, stream);
    conv_launch<384, 16>(Ebuf, wFn, b_final, finout, nullptr, 3, 0, stream);

    // out = base + out1 + out2
    pconv_apply<<<dim3((4 * 3 * 384 * 384 + 255) / 256), dim3(256), 0, stream>>>(slab, kout, outp);
    add_final<<<dim3((4 * 3 * 384 * 384 + 255) / 256), dim3(256), 0, stream>>>(finout, outp);
}
